// PGLSTM_3633542332574
// MI455X (gfx1250) — matrix operand order repaired and checked by reading, not compiled, not run on hardware
//
#include <hip/hip_runtime.h>
#include <cstdint>

// ---------------------------------------------------------------------------
// Types for CDNA5 WMMA (wave32): bf16 A/B fragments, f32 accumulators.
// ---------------------------------------------------------------------------
typedef __attribute__((ext_vector_type(16))) __bf16 v16bf;
typedef __attribute__((ext_vector_type(8)))  __bf16 v8bf;
typedef __attribute__((ext_vector_type(8)))  float  v8f;

#define B_   32
#define T_   512
#define H_   512
#define G_   2048          // 4*H
#define TB_  (T_ * B_)     // 16384 rows of xp
#define NT_  (G_ / 16)     // 128 N tiles
#define OUT_ELEMS  ((size_t)B_ * T_ * 2 * H_)   // 16,777,216
#define HN_ELEMS   ((size_t)4 * B_ * H_)        // 65,536

static __device__ __forceinline__ __bf16 f2bf(float f) {
  union { float f; uint32_t u; } v; v.f = f;
  uint32_t r = v.u + 0x7FFFu + ((v.u >> 16) & 1u);   // round-to-nearest-even
  unsigned short h = (unsigned short)(r >> 16);
  return __builtin_bit_cast(__bf16, h);
}

static __device__ __forceinline__ float sigf(float x) {
  return 1.0f / (1.0f + __expf(-x));
}
static __device__ __forceinline__ float tanh_fast(float x) {
  float e = __expf(-2.0f * x);
  return (1.0f - e) / (1.0f + e);
}

static __device__ __forceinline__ v8f wmma_bf16(v16bf a, v16bf b, v8f c) {
  // D = A(16x32) * B(32x16) + C, f32 accumulate
  return __builtin_amdgcn_wmma_f32_16x16x32_bf16(false, a, false, b,
                                                 (short)0, c, false, false);
}

static __device__ __forceinline__ v16bf cat8(v8bf lo, v8bf hi) {
  return __builtin_shufflevector(lo, hi, 0,1,2,3,4,5,6,7,8,9,10,11,12,13,14,15);
}

// Permuted G index: g' = (hb*4 + gate)*16 + col  <->  natural g = gate*512 + hb*16 + col.
static __device__ __forceinline__ int perm_to_nat(int gp) {
  int tile = gp >> 4, col = gp & 15;
  int gate = tile & 3, hb = tile >> 2;
  return gate * H_ + hb * 16 + col;
}

// ---------------------------------------------------------------------------
// 1) Hypernetwork weight generation:  W'[g'][k] = bf16( sum_m w[g][k][m]*d[m] )
//    with gate-interleaved row permutation on the G axis.
// ---------------------------------------------------------------------------
__global__ void gen_w_kernel(const float* __restrict__ w,   // (G, Din, 16)
                             const float* __restrict__ emb, // (8, 16)
                             const int*   __restrict__ dom,
                             __bf16* __restrict__ Wout,     // (G, Din) permuted rows
                             int Din) {
  size_t idx = (size_t)blockIdx.x * 256 + threadIdx.x;
  size_t total = (size_t)G_ * Din;
  if (idx >= total) return;
  int gp = (int)(idx / Din);
  int k  = (int)(idx % Din);
  int g  = perm_to_nat(gp);
  const float* d  = emb + dom[0] * 16;
  const float* wp = w + ((size_t)g * Din + k) * 16;
  float s = 0.f;
#pragma unroll
  for (int m = 0; m < 16; ++m) s += wp[m] * d[m];
  Wout[(size_t)gp * Din + k] = f2bf(s);
}

__global__ void gen_b_kernel(const float* __restrict__ bih, // (G,16)
                             const float* __restrict__ bhh, // (G,16)
                             const float* __restrict__ emb,
                             const int*   __restrict__ dom,
                             float* __restrict__ bout) {    // (G) permuted
  int gp = blockIdx.x * 256 + threadIdx.x;
  if (gp >= G_) return;
  int g = perm_to_nat(gp);
  const float* d = emb + dom[0] * 16;
  float s = 0.f;
#pragma unroll
  for (int m = 0; m < 16; ++m) s += (bih[g*16 + m] + bhh[g*16 + m]) * d[m];
  bout[gp] = s;
}

// ---------------------------------------------------------------------------
// 2) Cast + transpose x: (B,T,D) f32 -> (T*B, D) bf16 row-major (A matrix).
// ---------------------------------------------------------------------------
__global__ void cast_x_kernel(const float* __restrict__ x, __bf16* __restrict__ out) {
  size_t idx = (size_t)blockIdx.x * 256 + threadIdx.x;   // over B*T*512
  if (idx >= (size_t)B_ * T_ * 512) return;
  int k = (int)(idx & 511);
  size_t r = idx >> 9;
  int t = (int)(r & 511);
  int b = (int)(r >> 9);
  out[((size_t)t * B_ + b) * 512 + k] = f2bf(x[idx]);
}

// ---------------------------------------------------------------------------
// 3) Input-projection GEMM: xp = A(TBxK,bf16) * W(GxK,bf16)^T + bias, f32 out
//    in fragment-native tiled layout: xp[(mtile*128 + ntile)*256 + lane*8 + e].
//    Block = 256 thr (8 waves); wave tile = 32x32 (2x2 WMMA accums);
//    block tile = 128M x 64N.
// ---------------------------------------------------------------------------
__global__ __launch_bounds__(256) void xp_gemm_kernel(
    const __bf16* __restrict__ A, const __bf16* __restrict__ W,
    const float* __restrict__ bias, float* __restrict__ xp, int K) {
  const int wave  = threadIdx.x >> 5;
  const int lane  = threadIdx.x & 31;
  const int colN  = lane & 15;
  const int khalf = lane >> 4;
  const int mtile0 = blockIdx.y * 8 + (wave >> 1) * 2;
  const int ntile0 = blockIdx.x * 4 + (wave & 1) * 2;

  v8f acc[2][2] = {};
  const int kiter = K >> 5;
  for (int kk = 0; kk < kiter; ++kk) {
    v16bf a[2], b[2];
#pragma unroll
    for (int i = 0; i < 2; ++i) {
      const __bf16* ap = A + (size_t)((mtile0 + i) * 16 + colN) * K + kk * 32 + khalf * 16;
      a[i] = cat8(*(const v8bf*)ap, *(const v8bf*)(ap + 8));
      const __bf16* bp = W + (size_t)((ntile0 + i) * 16 + colN) * K + kk * 32 + khalf * 16;
      b[i] = *(const v16bf*)bp;
    }
#pragma unroll
    for (int mi = 0; mi < 2; ++mi)
#pragma unroll
      for (int ni = 0; ni < 2; ++ni)
        acc[mi][ni] = wmma_bf16(a[mi], b[ni], acc[mi][ni]);
  }
#pragma unroll
  for (int ni = 0; ni < 2; ++ni) {
    float bs = bias[(ntile0 + ni) * 16 + colN];
#pragma unroll
    for (int mi = 0; mi < 2; ++mi)
#pragma unroll
      for (int e = 0; e < 8; ++e) acc[mi][ni][e] += bs;
  }
#pragma unroll
  for (int mi = 0; mi < 2; ++mi)
#pragma unroll
    for (int ni = 0; ni < 2; ++ni) {
      size_t base = ((size_t)(mtile0 + mi) * NT_ + (ntile0 + ni)) * 256 + lane * 8;
      *(v8f*)(xp + base) = acc[mi][ni];
    }
}

// ---------------------------------------------------------------------------
// 4) Persistent LSTM scan: one 1024-thread block (32 waves) per direction.
//    Wave hb owns hidden block hb (16 units): 2(M) x 4(gate) WMMA accums.
//    h: bf16 in LDS (A-matrix friendly). c: resident in VGPR fragments.
//    Next-step xp fragments are prefetched (global_prefetch_b8) while the
//    current step's recurrent GEMM runs, hiding HBM latency on the serial
//    dependence chain (xp buffers are 268 MB total > 192 MB L2).
// ---------------------------------------------------------------------------
__global__ __launch_bounds__(1024, 1) void lstm_scan_kernel(
    const float*  __restrict__ xp0,      // dir0 tiled xp
    const float*  __restrict__ xp1,      // dir1 tiled xp
    const __bf16* __restrict__ whh,      // (2, G, H) bf16, permuted G rows
    __bf16* __restrict__ seq_bf16,       // layer0: (T*B, 2H) bf16 out, else null
    float*  __restrict__ out_f32,        // layer1: (B, T, 2H) f32 out, else null
    float*  __restrict__ hn,             // (4, B, H) region of d_out
    float*  __restrict__ cn,             // (4, B, H) region of d_out
    int layer) {
  const int dir  = blockIdx.x;
  const int hb   = threadIdx.x >> 5;     // 0..31 hidden block
  const int lane = threadIdx.x & 31;
  const int colN = lane & 15;
  const int khalf = lane >> 4;
  const float*  xp = dir ? xp1 : xp0;
  const __bf16* W  = whh + (size_t)dir * G_ * H_;

  __shared__ __bf16 hsh[B_][H_];         // 32 KB, row-major [b][j]
  for (int i = threadIdx.x; i < B_ * H_; i += 1024)
    (&hsh[0][0])[i] = f2bf(0.0f);
  __syncthreads();

  v8f cfrag[2] = {};                     // cell state, register resident
  v8f hfrag[2] = {};
  const int j = hb * 16 + colN;          // hidden unit column (natural order)

  for (int s = 0; s < T_; ++s) {
    const int t = dir ? (T_ - 1 - s) : s;

    // C init = xp_t (tiled fragment layout -> contiguous 32B per lane)
    v8f acc[2][4];
#pragma unroll
    for (int mt = 0; mt < 2; ++mt)
#pragma unroll
      for (int g = 0; g < 4; ++g) {
        size_t base = ((size_t)(2 * t + mt) * NT_ + (hb * 4 + g)) * 256 + lane * 8;
        acc[mt][g] = *(const v8f*)(xp + base);
      }

    // Prefetch next step's xp fragments (uniform branch; overlaps with GEMM)
    if (s + 1 < T_) {
      const int tn = dir ? (t - 1) : (t + 1);
#pragma unroll
      for (int mt = 0; mt < 2; ++mt)
#pragma unroll
        for (int g = 0; g < 4; ++g) {
          size_t base = ((size_t)(2 * tn + mt) * NT_ + (hb * 4 + g)) * 256 + lane * 8;
          __builtin_prefetch(xp + base, 0, 3);
        }
    }

    // g += h(t-1) @ W_hh^T   (K = 512)
    for (int kk = 0; kk < 16; ++kk) {
      v16bf a[2];
#pragma unroll
      for (int mt = 0; mt < 2; ++mt) {
        const __bf16* ap = &hsh[mt * 16 + colN][kk * 32 + khalf * 16];
        a[mt] = cat8(*(const v8bf*)ap, *(const v8bf*)(ap + 8));
      }
#pragma unroll
      for (int g = 0; g < 4; ++g) {
        const __bf16* bp = W + (size_t)((hb * 4 + g) * 16 + colN) * H_ + kk * 32 + khalf * 16;
        v16bf b = *(const v16bf*)bp;
        acc[0][g] = wmma_bf16(a[0], b, acc[0][g]);
        acc[1][g] = wmma_bf16(a[1], b, acc[1][g]);
      }
    }
    __syncthreads();   // all reads of h(t-1) complete before overwrite

    // Gates + state update, fully in fragment registers
#pragma unroll
    for (int mt = 0; mt < 2; ++mt) {
#pragma unroll
      for (int e = 0; e < 8; ++e) {
        float iv = sigf(acc[mt][0][e]);
        float fv = sigf(acc[mt][1][e]);
        float gv = tanh_fast(acc[mt][2][e]);
        float ov = sigf(acc[mt][3][e]);
        float cc = fv * cfrag[mt][e] + iv * gv;
        cfrag[mt][e] = cc;
        hfrag[mt][e] = ov * tanh_fast(cc);
      }
      // scatter h: LDS (next-step A) + sequence output
#pragma unroll
      for (int e = 0; e < 8; ++e) {
        int brow = mt * 16 + e + khalf * 8;           // batch index (C/D layout)
        __bf16 hb16 = f2bf(hfrag[mt][e]);
        hsh[brow][j] = hb16;
        if (seq_bf16)
          seq_bf16[((size_t)t * B_ + brow) * (2 * H_) + dir * H_ + j] = hb16;
        if (out_f32)
          out_f32[(size_t)brow * (T_ * 2 * H_) + (size_t)t * (2 * H_) + dir * H_ + j] =
              hfrag[mt][e];
      }
    }
    __syncthreads();   // h(t) visible to all waves before next GEMM
  }

  // final states h_n / c_n (natural hidden order)
  const size_t base = (size_t)(layer * 2 + dir) * B_ * H_;
#pragma unroll
  for (int mt = 0; mt < 2; ++mt)
#pragma unroll
    for (int e = 0; e < 8; ++e) {
      int brow = mt * 16 + e + khalf * 8;
      hn[base + (size_t)brow * H_ + j] = hfrag[mt][e];
      cn[base + (size_t)brow * H_ + j] = cfrag[mt][e];
    }
}

// ---------------------------------------------------------------------------
// Host orchestration
// ---------------------------------------------------------------------------
extern "C" void kernel_launch(void* const* d_in, const int* in_sizes, int n_in,
                              void* d_out, int out_size, void* d_ws, size_t ws_size,
                              hipStream_t stream) {
  (void)in_sizes; (void)n_in; (void)out_size; (void)ws_size;
  const float* x      = (const float*)d_in[0];
  const float* emb    = (const float*)d_in[1];
  const float* w_ih_0 = (const float*)d_in[2];
  const float* w_hh_0 = (const float*)d_in[3];
  const float* b_ih_0 = (const float*)d_in[4];
  const float* b_hh_0 = (const float*)d_in[5];
  const float* w_ih_1 = (const float*)d_in[6];
  const float* w_hh_1 = (const float*)d_in[7];
  const float* b_ih_1 = (const float*)d_in[8];
  const float* b_hh_1 = (const float*)d_in[9];
  const int*   dom    = (const int*)d_in[10];
  float* out = (float*)d_out;

  // workspace carve-up (256B aligned)
  char* ws = (char*)d_ws;
  size_t off = 0;
  auto carve = [&](size_t bytes) -> char* {
    char* p = ws + off;
    off += (bytes + 255) & ~(size_t)255;
    return p;
  };
  __bf16* wih0 = (__bf16*)carve((size_t)2 * G_ * 512  * 2);
  __bf16* whh0 = (__bf16*)carve((size_t)2 * G_ * 512  * 2);
  __bf16* wih1 = (__bf16*)carve((size_t)2 * G_ * 1024 * 2);
  __bf16* whh1 = (__bf16*)carve((size_t)2 * G_ * 512  * 2);
  float*  bias = (float*)carve((size_t)4 * G_ * 4);       // [layer*2+dir][G]
  __bf16* xbf  = (__bf16*)carve((size_t)TB_ * 512  * 2);
  __bf16* seq0 = (__bf16*)carve((size_t)TB_ * 1024 * 2);
  float*  xpA  = (float*)carve((size_t)TB_ / 16 * NT_ * 256 * 4);  // 134 MB
  float*  xpB  = (float*)carve((size_t)TB_ / 16 * NT_ * 256 * 4);

  float* hn = out + OUT_ELEMS;
  float* cn = out + OUT_ELEMS + HN_ELEMS;

  // --- 1) weight / bias generation (permuted G, bf16) ---
  for (int d = 0; d < 2; ++d) {
    int nb512  = (int)(((size_t)G_ * 512  + 255) / 256);
    int nb1024 = (int)(((size_t)G_ * 1024 + 255) / 256);
    gen_w_kernel<<<nb512, 256, 0, stream>>>(
        w_ih_0 + (size_t)d * G_ * 512 * 16, emb, dom, wih0 + (size_t)d * G_ * 512, 512);
    gen_w_kernel<<<nb512, 256, 0, stream>>>(
        w_hh_0 + (size_t)d * G_ * 512 * 16, emb, dom, whh0 + (size_t)d * G_ * 512, 512);
    gen_w_kernel<<<nb1024, 256, 0, stream>>>(
        w_ih_1 + (size_t)d * G_ * 1024 * 16, emb, dom, wih1 + (size_t)d * G_ * 1024, 1024);
    gen_w_kernel<<<nb512, 256, 0, stream>>>(
        w_hh_1 + (size_t)d * G_ * 512 * 16, emb, dom, whh1 + (size_t)d * G_ * 512, 512);
    gen_b_kernel<<<G_ / 256, 256, 0, stream>>>(
        b_ih_0 + (size_t)d * G_ * 16, b_hh_0 + (size_t)d * G_ * 16, emb, dom,
        bias + (size_t)(0 * 2 + d) * G_);
    gen_b_kernel<<<G_ / 256, 256, 0, stream>>>(
        b_ih_1 + (size_t)d * G_ * 16, b_hh_1 + (size_t)d * G_ * 16, emb, dom,
        bias + (size_t)(1 * 2 + d) * G_);
  }

  // --- 2) cast/transpose x -> (T*B, 512) bf16 ---
  {
    int nb = (int)(((size_t)B_ * T_ * 512 + 255) / 256);
    cast_x_kernel<<<nb, 256, 0, stream>>>(x, xbf);
  }

  // --- 3) layer 0: xp GEMMs then scan ---
  dim3 ggrid(G_ / 64, TB_ / 128);
  xp_gemm_kernel<<<ggrid, 256, 0, stream>>>(xbf, wih0,            bias + 0 * G_, xpA, 512);
  xp_gemm_kernel<<<ggrid, 256, 0, stream>>>(xbf, wih0 + (size_t)G_ * 512,
                                            bias + 1 * G_, xpB, 512);
  lstm_scan_kernel<<<dim3(2), 1024, 0, stream>>>(xpA, xpB, whh0, seq0, nullptr,
                                                 hn, cn, 0);

  // --- 4) layer 1: xp GEMMs (K=1024) then scan writing final output ---
  xp_gemm_kernel<<<ggrid, 256, 0, stream>>>(seq0, wih1,            bias + 2 * G_, xpA, 1024);
  xp_gemm_kernel<<<ggrid, 256, 0, stream>>>(seq0, wih1 + (size_t)G_ * 1024,
                                            bias + 3 * G_, xpB, 1024);
  lstm_scan_kernel<<<dim3(2), 1024, 0, stream>>>(xpA, xpB, whh1, nullptr, out,
                                                 hn, cn, 1);
}
